// karklin_lewicki_65532611002701
// MI455X (gfx1250) — compile-verified, matrix operand order checked by reading, not run on hardware
//
#include <hip/hip_runtime.h>

// ---------------------------------------------------------------------------
// Karklin-Lewicki sparse coding, 19 gradient steps on MI455X (gfx1250).
// 4 GEMMs/step as bf16 WMMA (v_wmma_f32_16x16x32_bf16), f32 accumulate,
// fused epilogues. bf16 activation mirrors stored N-major so every GEMM
// operand stages into LDS with contiguous 64B transfers. Double-buffered
// LDS with GLOBAL_LOAD_ASYNC_TO_LDS_B128 (ASYNCcnt) compute/copy overlap.
// Working set (~60MB) is L2-resident (192MB); compute-bound -> matrix cores.
// ---------------------------------------------------------------------------

typedef __attribute__((ext_vector_type(16))) __bf16 v16bf;
typedef __attribute__((ext_vector_type(8)))  float  v8f;
typedef __attribute__((ext_vector_type(4)))  int    v4i;
typedef unsigned short bfraw;

// AS-qualified pointers for the async-LDS builtin:
// param0: v4i in addrspace(1) (global), param1: v4i in addrspace(3) (LDS)
typedef v4i __attribute__((address_space(1)))* v4i_gp;
typedef v4i __attribute__((address_space(3)))* v4i_lp;

#define NPIX 1024
#define NU   4096
#define NV   1024
#define NB   512
#define INVB (1.0f / 512.0f)
#define NSTEPS 19

#if defined(__has_builtin)
#if __has_builtin(__builtin_amdgcn_global_load_async_to_lds_b128)
#define HAVE_ASYNC_LDS 1
#endif
#endif
#ifndef HAVE_ASYNC_LDS
#define HAVE_ASYNC_LDS 0
#endif

static __device__ __forceinline__ bfraw f2bf(float f) {
  unsigned int u = __float_as_uint(f);
  unsigned int r = 0x7FFFu + ((u >> 16) & 1u);   // round-to-nearest-even
  return (bfraw)((u + r) >> 16);
}
static __device__ __forceinline__ unsigned int pack2bf(float lo, float hi) {
  return (unsigned int)f2bf(lo) | ((unsigned int)f2bf(hi) << 16);
}

// 16B global -> LDS copy: async DMA path (no VGPR round-trip, ASYNCcnt)
static __device__ __forceinline__ void cp16(const void* g, void* l) {
#if HAVE_ASYNC_LDS
  __builtin_amdgcn_global_load_async_to_lds_b128((v4i_gp)g, (v4i_lp)l, 0, 0);
#else
  *(uint4*)l = *(const uint4*)g;
#endif
}
static __device__ __forceinline__ void wait_cp(bool more_in_flight) {
#if HAVE_ASYNC_LDS
#if defined(__has_builtin) && __has_builtin(__builtin_amdgcn_s_wait_asynccnt)
  if (more_in_flight) __builtin_amdgcn_s_wait_asynccnt(8);
  else                __builtin_amdgcn_s_wait_asynccnt(0);
#else
  if (more_in_flight) asm volatile("s_wait_asynccnt 0x8" ::: "memory");
  else                asm volatile("s_wait_asynccnt 0x0" ::: "memory");
#endif
#else
  (void)more_in_flight;   // sync path: compiler inserts loadcnt/dscnt waits
#endif
}

// f32 weight [R,C] -> bf16 copy + bf16 transpose (one-time, L2-resident after)
__global__ __launch_bounds__(256)
void convert_wt(const float* __restrict__ src, bfraw* __restrict__ dst,
                bfraw* __restrict__ dstT, int R, int C) {
  int idx = blockIdx.x * 256 + threadIdx.x;   // R*C multiple of 256
  int r = idx / C, c = idx % C;
  bfraw h = f2bf(src[idx]);
  dst[idx] = h;
  dstT[(size_t)c * R + r] = h;
}

// ---------------------------------------------------------------------------
// acc[M,N] = A[M,K] (row-major, lda=K) x Bt[N,K] (n-major, ld=K)
//  MODE 1: s=exp(acc)->o0[M,N];  gT=bf16(u*s-1)->o1[N,M]      (e0=u)
//  MODE 2: rT=bf16(acc-x)->o1[N,M]                            (e0=x)
//  MODE 3: u -= (recon/B)*acc + (1/B)*s; o0=u[M,N] rw, o1=uT  (e0=s)
//  MODE 4: v -= (1/B)*acc + (sparse/B)*sign(v); o0=v rw, o1=vT
// Block 256 thr (8 wave32), tile 128x128, BK=64, double-buffered LDS.
// Wave tile 64x32 = 4x2 fragments; 16 WMMAs per K-tile per wave.
// ---------------------------------------------------------------------------
template<int MODE>
__global__ __launch_bounds__(256)
void gemm_step(const bfraw* __restrict__ A,
               const bfraw* __restrict__ Bt,
               const float* __restrict__ e0,
               float* __restrict__ o0,
               bfraw* __restrict__ o1T,
               const float* __restrict__ scal,
               int M, int N, int K) {
  __shared__ bfraw As[2][128][72];   // [buf][m][k], +8 pad (144B rows, 16B aligned)
  __shared__ bfraw Bs[2][128][72];   // [buf][n][k]

  const int tid    = threadIdx.x;
  const int lane   = tid & 31;
  const int wave   = tid >> 5;
  const int wm     = wave >> 2;          // 0..1
  const int wn     = wave & 3;           // 0..3
  const int mBlock = blockIdx.y * 128;
  const int nBlock = blockIdx.x * 128;
  const int lanelo = lane & 15;
  const bool hi    = lane >= 16;
  const int koff   = hi ? 8 : 0;         // A/B frag lane->K map (ISA 7.12.2)

  v8f zero = {};
  v8f acc[4][2];
#pragma unroll
  for (int i = 0; i < 4; ++i)
#pragma unroll
    for (int j = 0; j < 2; ++j) acc[i][j] = zero;

  const int row   = tid >> 1;            // 0..127 staging row
  const int halfk = (tid & 1) * 32;      // 0 / 32

  // issue 8x16B transfers for one K-tile (4 for A, 4 for B) into buffer `buf`
  auto stage = [&](int buf, int kt) {
    const bfraw* ag = A  + (size_t)(mBlock + row) * K + kt + halfk;
    const bfraw* bg = Bt + (size_t)(nBlock + row) * K + kt + halfk;
#if !HAVE_ASYNC_LDS
    if (kt + 64 < K) {
      __builtin_prefetch(ag + 64, 0, 1);
      __builtin_prefetch(bg + 64, 0, 1);
    }
#endif
#pragma unroll
    for (int q = 0; q < 4; ++q) {
      cp16(ag + q * 8, &As[buf][row][halfk + q * 8]);
      cp16(bg + q * 8, &Bs[buf][row][halfk + q * 8]);
    }
  };

  const int nTiles = K >> 6;
  stage(0, 0);
  for (int t = 0; t < nTiles; ++t) {
    const int cur = t & 1;
    const bool more = (t + 1) < nTiles;
    if (more) stage(cur ^ 1, (t + 1) << 6);   // overlaps with compute below
    wait_cp(more);                            // current tile's 8 xfers landed
    __syncthreads();

#pragma unroll
    for (int kc = 0; kc < 64; kc += 32) {
      union FragU { uint4 q[2]; v16bf v; };
      FragU af[4], bfrg[2];
#pragma unroll
      for (int i = 0; i < 4; ++i) {
        const bfraw* p = &As[cur][wm * 64 + i * 16 + lanelo][kc + koff];
        af[i].q[0] = *(const uint4*)p;
        af[i].q[1] = *(const uint4*)(p + 16);
      }
#pragma unroll
      for (int j = 0; j < 2; ++j) {
        const bfraw* p = &Bs[cur][wn * 32 + j * 16 + lanelo][kc + koff];
        bfrg[j].q[0] = *(const uint4*)p;
        bfrg[j].q[1] = *(const uint4*)(p + 16);
      }
#pragma unroll
      for (int i = 0; i < 4; ++i)
#pragma unroll
        for (int j = 0; j < 2; ++j)
          acc[i][j] = __builtin_amdgcn_wmma_f32_16x16x32_bf16(
              false, af[i].v, false, bfrg[j].v, (short)0, acc[i][j],
              false, false);
    }
    __syncthreads();   // all waves done reading `cur` before it is re-staged
  }

  // ---- fused epilogue -----------------------------------------------------
  // C/D layout: VGPR r holds (m = base + hi*8 + r, n = base + lanelo); a lane
  // owns 8 consecutive m for one n -> one 16B store to the N-major mirror.
  float sc = (MODE == 3 || MODE == 4) ? scal[0] : 0.0f;
#pragma unroll
  for (int i = 0; i < 4; ++i) {
#pragma unroll
    for (int j = 0; j < 2; ++j) {
      const int mf = mBlock + wm * 64 + i * 16 + (hi ? 8 : 0);
      const int n  = nBlock + wn * 32 + j * 16 + lanelo;
      float outv[8];
#pragma unroll
      for (int r = 0; r < 8; ++r) {
        size_t idx = (size_t)(mf + r) * N + n;
        float av = acc[i][j][r];
        if (MODE == 1) {                 // s = exp(b@v); g = u*s - 1
          float sv = __expf(av);
          o0[idx] = sv;
          outv[r] = e0[idx] * sv - 1.0f;
        } else if (MODE == 2) {          // residual = a@u - x
          outv[r] = av - e0[idx];
        } else if (MODE == 3) {          // u update
          float un = o0[idx] - (sc * INVB) * av - INVB * e0[idx];
          o0[idx] = un;
          outv[r] = un;
        } else {                         // v update
          float vv = o0[idx];
          float sg = (vv > 0.0f) ? 1.0f : ((vv < 0.0f) ? -1.0f : 0.0f);
          float vn = vv - INVB * av - (sc * INVB) * sg;
          o0[idx] = vn;
          outv[r] = vn;
        }
      }
      uint4 pk;
      pk.x = pack2bf(outv[0], outv[1]);
      pk.y = pack2bf(outv[2], outv[3]);
      pk.z = pack2bf(outv[4], outv[5]);
      pk.w = pack2bf(outv[6], outv[7]);
      *(uint4*)(o1T + (size_t)n * M + mf) = pk;   // [N][M] mirror, 16B store
    }
  }
}

// ---------------------------------------------------------------------------
extern "C" void kernel_launch(void* const* d_in, const int* in_sizes, int n_in,
                              void* d_out, int out_size, void* d_ws, size_t ws_size,
                              hipStream_t stream) {
  (void)in_sizes; (void)n_in; (void)out_size; (void)ws_size;

  const float* x        = (const float*)d_in[0];   // [NPIX, NB]
  const float* a        = (const float*)d_in[1];   // [NPIX, NU]
  const float* b        = (const float*)d_in[2];   // [NU, NV]
  const float* recon_p  = (const float*)d_in[3];   // scalar (device)
  const float* sparse_p = (const float*)d_in[4];   // scalar (device)

  float* u = (float*)d_out;                        // [NU, NB]
  float* v = u + (size_t)NU * NB;                  // [NV, NB]

  char* ws = (char*)d_ws;
  size_t off = 0;
  auto take = [&](size_t bytes) -> char* {
    char* p = ws + off;
    off += (bytes + 255) & ~(size_t)255;
    return p;
  };
  bfraw* a_bf  = (bfraw*)take((size_t)NPIX * NU * 2);   // a  [NPIX,NU]
  bfraw* aT_bf = (bfraw*)take((size_t)NPIX * NU * 2);   // aT [NU,NPIX]
  bfraw* b_bf  = (bfraw*)take((size_t)NU * NV * 2);     // b  [NU,NV]
  bfraw* bT_bf = (bfraw*)take((size_t)NU * NV * 2);     // bT [NV,NU]
  bfraw* uT_bf = (bfraw*)take((size_t)NU * NB * 2);     // u mirror [NB,NU]
  bfraw* vT_bf = (bfraw*)take((size_t)NV * NB * 2);     // v mirror [NB,NV]
  bfraw* rT_bf = (bfraw*)take((size_t)NPIX * NB * 2);   // r mirror [NB,NPIX]
  bfraw* gT_bf = (bfraw*)take((size_t)NU * NB * 2);     // g mirror [NB,NU]
  float* s_buf = (float*)take((size_t)NU * NB * 4);     // exp(b@v) [NU,NB]

  convert_wt<<<(NPIX * NU) / 256, 256, 0, stream>>>(a, a_bf, aT_bf, NPIX, NU);
  convert_wt<<<(NU * NV) / 256, 256, 0, stream>>>(b, b_bf, bT_bf, NU, NV);

  (void)hipMemsetAsync(d_out, 0, (size_t)(NU + NV) * NB * sizeof(float), stream);
  (void)hipMemsetAsync(uT_bf, 0, (size_t)NU * NB * 2, stream);
  (void)hipMemsetAsync(vT_bf, 0, (size_t)NV * NB * 2, stream);

  dim3 blk(256);
  dim3 gBig(NB / 128, NU / 128);      // 4 x 32  (M = 4096)
  dim3 gSmall(NB / 128, NPIX / 128);  // 4 x 8   (M = 1024)

  for (int step = 0; step < NSTEPS; ++step) {
    // 1) bv = b@v ; s = exp(bv) ; gT = bf16(u*s - 1)      [OLD u, v]
    gemm_step<1><<<gBig, blk, 0, stream>>>(b_bf, vT_bf, u, s_buf, gT_bf,
                                           recon_p, NU, NB, NV);
    // 2) rT = bf16(a@u - x)                               [OLD u]
    gemm_step<2><<<gSmall, blk, 0, stream>>>(a_bf, uT_bf, x, nullptr, rT_bf,
                                             recon_p, NPIX, NB, NU);
    // 3) u -= (recon/B)*aT@r + (1/B)*s ; refresh uT_bf
    gemm_step<3><<<gBig, blk, 0, stream>>>(aT_bf, rT_bf, s_buf, u, uT_bf,
                                           recon_p, NU, NB, NPIX);
    // 4) v -= (1/B)*bT@g + (sparse/B)*sign(v) ; refresh vT_bf
    gemm_step<4><<<gSmall, blk, 0, stream>>>(bT_bf, gT_bf, nullptr, v, vT_bf,
                                             sparse_p, NV, NB, NU);
  }
}
